// CharRNN_618475291355
// MI455X (gfx1250) — matrix operand order repaired and checked by reading, not compiled, not run on hardware
//
#include <hip/hip_runtime.h>
#include <hip/hip_bf16.h>

typedef __attribute__((ext_vector_type(16))) __bf16 v16bf;
typedef __attribute__((ext_vector_type(8)))  float  v8f;

constexpr int kB = 256;    // batch
constexpr int kT = 512;    // seq len
constexpr int kH = 256;    // hidden
constexpr int kV = 128;    // vocab
constexpr int kL = 2;      // layers
constexpr int kG = 4 * kH;     // 1024 gate width
constexpr int kK = 2 * kH;     // 512 = concat(x_t, h) K-dim
constexpr int kKSteps = kK / 32;  // 16 WMMA K-steps
constexpr int kRowPad = 528;   // LDS row stride in bf16 (32B-aligned, bank-spread)

__device__ __forceinline__ float sigmf(float x) { return 1.f / (1.f + __expf(-x)); }

__device__ __forceinline__ v8f wmma_bf16(v16bf a, v16bf b, v8f c) {
  return __builtin_amdgcn_wmma_f32_16x16x32_bf16(false, a, false, b,
                                                 (short)0, c, false, false);
}

// ---------------------------------------------------------------------------
// Embedding lookup: x[B,T] int32 -> emb[T,B,H] bf16 (time-major for the LSTM)
// ---------------------------------------------------------------------------
__global__ void embed_kernel(const int* __restrict__ x, const float* __restrict__ Wemb,
                             __bf16* __restrict__ emb) {
  size_t idx = (size_t)blockIdx.x * blockDim.x + threadIdx.x;  // < T*B*H
  int t   = (int)(idx / (kB * kH));
  int rem = (int)(idx % (kB * kH));
  int b = rem >> 8;
  int h = rem & (kH - 1);
  int xi = x[(size_t)b * kT + t];
  emb[idx] = (__bf16)Wemb[(size_t)xi * kH + h];
}

// ---------------------------------------------------------------------------
// Weight repack: W_ih/W_hh [L,4H,H] fp32 -> Wt [L, N=1024, K=512] bf16 (K-major
// so the WMMA B-fragment loads are contiguous); fc_w -> [N=128, K=256] bf16;
// bsum = b_ih + b_hh.
// ---------------------------------------------------------------------------
__global__ void pack_kernel(const float* __restrict__ Wih, const float* __restrict__ Whh,
                            const float* __restrict__ bih, const float* __restrict__ bhh,
                            const float* __restrict__ fcw,
                            __bf16* __restrict__ Wt, __bf16* __restrict__ fcwt,
                            float* __restrict__ bsum) {
  size_t idx = (size_t)blockIdx.x * blockDim.x + threadIdx.x;  // < 2*512*1024
  {
    int l = (int)(idx >> 19);
    int r = (int)(idx & ((1 << 19) - 1));
    int k = r >> 10;
    int n = r & (kG - 1);
    float v = (k < kH) ? Wih[((size_t)l * kG + n) * kH + k]
                       : Whh[((size_t)l * kG + n) * kH + (k - kH)];
    Wt[((size_t)l * kG + n) * kK + k] = (__bf16)v;
  }
  if (idx < (size_t)kH * kV) {
    int k = (int)(idx >> 7);
    int n = (int)(idx & (kV - 1));
    fcwt[(size_t)n * kH + k] = (__bf16)fcw[(size_t)n * kH + k];
  }
  if (idx < (size_t)kL * kG) bsum[idx] = bih[idx] + bhh[idx];
}

// ---------------------------------------------------------------------------
// Per-layer state init: h0/c0 -> hstate(bf16)/cstate(fp32); zero barrier ctr.
// ---------------------------------------------------------------------------
__global__ void init_kernel(const float* __restrict__ h0, const float* __restrict__ c0,
                            __bf16* __restrict__ hstate, float* __restrict__ cstate,
                            unsigned* __restrict__ cnt) {
  size_t idx = (size_t)blockIdx.x * blockDim.x + threadIdx.x;  // < B*H
  hstate[idx] = (__bf16)h0[idx];
  cstate[idx] = c0[idx];
  if (idx == 0) *cnt = 0u;
}

// ---------------------------------------------------------------------------
// Device-wide barrier for the persistent recurrence kernel (64 blocks).
// ---------------------------------------------------------------------------
__device__ __forceinline__ void grid_barrier(unsigned* cnt, unsigned target) {
  __syncthreads();
  __threadfence();
  if (threadIdx.x == 0) {
    atomicAdd(cnt, 1u);
    volatile unsigned* vc = (volatile unsigned*)cnt;
    while (*vc < target) { __builtin_amdgcn_s_sleep(2); }
  }
  __syncthreads();
  __threadfence();
}

// ---------------------------------------------------------------------------
// Persistent LSTM recurrence for one layer. Grid = 64 blocks x 256 threads
// (512 waves). Weights for this wave's 16x32 output slab are preloaded into
// 256 VGPRs of v16bf fragments and stay resident across all 512 timesteps
// (immune to the asm memory clobbers). Per step:
//   (a) async-copy this block's A tile [16 rows x K=512] of [x_t ; h] into
//       LDS (global_load_async_to_lds_b128, s_wait_asynccnt),
//   (b) 32x ds_load_b128 + 32x v_wmma_f32_16x16x32_bf16 on two independent
//       accumulator chains,
//   (c) grid barrier, elementwise gate math -> h/c update, grid barrier.
// ---------------------------------------------------------------------------
__global__ void __launch_bounds__(256, 1)
lstm_recur_kernel(const __bf16* __restrict__ xin,   // [T,B,H] layer input
                  __bf16* __restrict__ hout,        // [T,B,H] layer output
                  const __bf16* __restrict__ Wt,    // [N=1024, K=512]
                  const float* __restrict__ bsum,   // [1024]
                  __bf16* __restrict__ hstate,      // [B,H]
                  float* __restrict__ cstate,       // [B,H]
                  float* __restrict__ gates,        // [B,1024] fp32 scratch
                  unsigned* __restrict__ cnt,
                  float* __restrict__ out_hT,       // [B,H] fp32 (d_out slice)
                  float* __restrict__ out_cT) {     // [B,H] fp32 (d_out slice)
  __shared__ __align__(32) __bf16 sA[16 * kRowPad];  // 16 rows x 512 K (padded)

  const int tid  = threadIdx.x;
  const int lane = tid & 31;
  const int wid  = tid >> 5;
  const int w    = blockIdx.x * 8 + wid;       // 0..511 global wave id
  const int m0b  = (blockIdx.x >> 2) * 16;     // block-uniform M base
  const int n0   = (w & 31) * 32;              // 32 N-pairs (each 2 tiles)
  const int mrow = m0b + 8 * (lane >> 4);      // C-fragment row base
  const int ncol = lane & 15;                  // C-fragment col
  const int ahi  = (lane >> 4) << 4;           // A-fragment K-half select
  const unsigned ldsBase = (unsigned)(uintptr_t)&sA[0];
  unsigned nbar = 0;

  // Preload this wave's B fragments (weights) into registers: they are
  // loop-invariant across the whole recurrence.
  v16bf bw0[kKSteps], bw1[kKSteps];
#pragma unroll
  for (int kk = 0; kk < kKSteps; ++kk) {
    bw0[kk] = *(const v16bf*)(Wt + (size_t)(n0 + (lane & 15)) * kK + kk * 32 + ahi);
    bw1[kk] = *(const v16bf*)(Wt + (size_t)(n0 + 16 + (lane & 15)) * kK + kk * 32 + ahi);
  }

  const __bf16* arow = &sA[(lane & 15) * kRowPad + ahi];

  for (int t = 0; t < kT; ++t) {
    // ---- (a) Async stage A tile into LDS ----------------------------------
    // 1024 16-byte chunks: 16 rows x 64 chunks (x part cc<32, h part cc>=32).
    {
      const __bf16* xrow = xin + (size_t)t * kB * kH;
#pragma unroll
      for (int i = 0; i < 4; ++i) {
        int c   = tid + i * 256;
        int row = c >> 6;
        int cc  = c & 63;
        const __bf16* g = (cc < 32)
            ? (xrow   + (size_t)(m0b + row) * kH + cc * 8)
            : (hstate + (size_t)(m0b + row) * kH + (cc - 32) * 8);
        unsigned lo = ldsBase + (unsigned)(row * (kRowPad * 2) + cc * 16);
        unsigned long long ga = (unsigned long long)(uintptr_t)g;
        asm volatile("global_load_async_to_lds_b128 %0, %1, off"
                     :: "v"(lo), "v"(ga) : "memory");
      }
    }
    asm volatile("s_wait_asynccnt 0x0" ::: "memory");
    __syncthreads();

    // ---- (b) gate GEMM: LDS A-fragments x register-resident weights -------
    v8f acc0 = {};
    v8f acc1 = {};
#pragma unroll
    for (int kk = 0; kk < kKSteps; ++kk) {
      v16bf a = *(const v16bf*)(arow + kk * 32);
      acc0 = wmma_bf16(a, bw0[kk], acc0);
      acc1 = wmma_bf16(a, bw1[kk], acc1);
    }
#pragma unroll
    for (int r = 0; r < 8; ++r) {
      gates[(size_t)(mrow + r) * kG + n0 + ncol]      = acc0[r];
      gates[(size_t)(mrow + r) * kG + n0 + 16 + ncol] = acc1[r];
    }

    ++nbar;
    grid_barrier(cnt, nbar * 64u);

    // ---- (c) elementwise gate nonlinearity ---------------------------------
#pragma unroll
    for (int i = 0; i < 4; ++i) {
      int idx = (blockIdx.x * 256 + tid) + i * 16384;  // < B*H = 65536
      int br = idx >> 8;
      int j  = idx & (kH - 1);
      const float* gr = gates + (size_t)br * kG;
      float gi = gr[j]          + bsum[j];
      float gf = gr[kH + j]     + bsum[kH + j];
      float gg = gr[2 * kH + j] + bsum[2 * kH + j];
      float go = gr[3 * kH + j] + bsum[3 * kH + j];
      float c  = cstate[idx];
      float cn = sigmf(gf) * c + sigmf(gi) * tanhf(gg);
      float hn = sigmf(go) * tanhf(cn);
      cstate[idx] = cn;
      hstate[idx] = (__bf16)hn;
      hout[(size_t)t * kB * kH + idx] = (__bf16)hn;
      if (t == kT - 1) { out_hT[idx] = hn; out_cT[idx] = cn; }
    }

    // Prefetch next timestep's x rows for this block (global_prefetch_b8).
    if (t + 1 < kT && tid < 128) {
      const __bf16* p = xin + (size_t)(t + 1) * kB * kH
                      + (size_t)(m0b + (tid >> 3)) * kH + (size_t)(tid & 7) * 32;
      __builtin_prefetch(p, 0, 1);
    }

    ++nbar;
    grid_barrier(cnt, nbar * 64u);
  }
}

// ---------------------------------------------------------------------------
// Final FC: logits[B*T,128] = h[B*T,256] @ fc_w^T + fc_b, bf16 WMMA.
// h is time-major [T,B,H]; logits rows are b*T+t. One 16x16 tile per wave.
// ---------------------------------------------------------------------------
__global__ void __launch_bounds__(256)
fc_kernel(const __bf16* __restrict__ hin,   // [T,B,H]
          const __bf16* __restrict__ fcwt,  // [N=128, K=256]
          const float* __restrict__ fcb, float* __restrict__ logits) {
  const int lane = threadIdx.x & 31;
  const int wid  = threadIdx.x >> 5;
  const int w    = blockIdx.x * 8 + wid;   // 0..65535
  const int m0   = (w >> 3) * 16;
  const int n0   = (w & 7) * 16;
  const int mA   = m0 + (lane & 15);       // logits row for A fragment
  const int b    = mA >> 9;                // / T
  const int t    = mA & (kT - 1);
  const int ahi  = (lane >> 4) << 4;
  v8f acc = {};
  for (int k0 = 0; k0 < kH; k0 += 32) {
    v16bf a  = *(const v16bf*)(hin + ((size_t)t * kB + b) * kH + k0 + ahi);
    v16bf bm = *(const v16bf*)(fcwt + (size_t)(n0 + (lane & 15)) * kH + k0 + ahi);
    acc = wmma_bf16(a, bm, acc);
  }
  const int mrow = m0 + 8 * (lane >> 4);
  const int n    = n0 + (lane & 15);
  const float bias = fcb[n];
#pragma unroll
  for (int r = 0; r < 8; ++r)
    logits[(size_t)(mrow + r) * kV + n] = acc[r] + bias;
}

// ---------------------------------------------------------------------------
// Launch: pack + embed (parallel prep), per-layer init + persistent
// recurrence (layer-1 output aliases the embedding buffer), final FC GEMM.
// Workspace layout (bytes):
//   bufA   @ 0          64 MiB  bf16 [T,B,H]  emb / layer-1 output
//   bufB   @ 0x4000000  64 MiB  bf16 [T,B,H]  layer-0 output
//   Wt     @ 0x8000000   2 MiB  bf16 [L,1024,512]
//   fcwt   @ +2 MiB     64 KiB  bf16 [128,256]
//   bsum   @ ...         8 KiB  f32  [L,1024]
//   gates  @ ...         1 MiB  f32  [256,1024]
//   hstate @ ...       128 KiB  bf16 [256,256]
//   cstate @ ...       256 KiB  f32  [256,256]
//   cnt    @ ...         4 B    barrier counter
// ---------------------------------------------------------------------------
extern "C" void kernel_launch(void* const* d_in, const int* in_sizes, int n_in,
                              void* d_out, int out_size, void* d_ws, size_t ws_size,
                              hipStream_t stream) {
  const int*   x    = (const int*)d_in[0];
  const float* h0   = (const float*)d_in[1];
  const float* c0   = (const float*)d_in[2];
  const float* Wemb = (const float*)d_in[3];
  const float* Wih  = (const float*)d_in[4];
  const float* Whh  = (const float*)d_in[5];
  const float* bih  = (const float*)d_in[6];
  const float* bhh  = (const float*)d_in[7];
  const float* fcw  = (const float*)d_in[8];
  const float* fcb  = (const float*)d_in[9];
  float* out = (float*)d_out;

  char* ws = (char*)d_ws;
  __bf16*   bufA  = (__bf16*)(ws);
  __bf16*   bufB  = (__bf16*)(ws + 67108864);
  __bf16*   Wt    = (__bf16*)(ws + 134217728);
  __bf16*   fcwt  = (__bf16*)(ws + 136314880);
  float*    bsum  = (float*)(ws + 136380416);
  float*    gates = (float*)(ws + 136388608);
  __bf16*   hst   = (__bf16*)(ws + 137437184);
  float*    cst   = (float*)(ws + 137568256);
  unsigned* cnt   = (unsigned*)(ws + 137830400);

  dim3 blk(256);
  pack_kernel<<<4096, blk, 0, stream>>>(Wih, Whh, bih, bhh, fcw, Wt, fcwt, bsum);
  embed_kernel<<<131072, blk, 0, stream>>>(x, Wemb, bufA);

  float* out_h = out + (size_t)kB * kT * kV;          // hTs [L,B,H]
  float* out_c = out_h + (size_t)kL * kB * kH;        // cTs [L,B,H]

  // Layer 0: bufA -> bufB
  init_kernel<<<256, blk, 0, stream>>>(h0, c0, hst, cst, cnt);
  lstm_recur_kernel<<<64, blk, 0, stream>>>(bufA, bufB, Wt, bsum, hst, cst,
                                            gates, cnt, out_h, out_c);
  // Layer 1: bufB -> bufA (emb no longer needed)
  init_kernel<<<256, blk, 0, stream>>>(h0 + (size_t)kB * kH, c0 + (size_t)kB * kH,
                                       hst, cst, cnt);
  lstm_recur_kernel<<<64, blk, 0, stream>>>(bufB, bufA, Wt + (size_t)kK * kG,
                                            bsum + kG, hst, cst, gates, cnt,
                                            out_h + (size_t)kB * kH,
                                            out_c + (size_t)kB * kH);

  fc_kernel<<<8192, blk, 0, stream>>>(bufA, fcwt, fcb, out);
}
